// Relation_module_72129680769734
// MI455X (gfx1250) — matrix operand order repaired and checked by reading, not compiled, not run on hardware
//
#include <hip/hip_runtime.h>
#include <cstdint>
#include <cstddef>

// ---------------------------------------------------------------------------
// Relation module for MI455X (gfx1250, wave32, WMMA).
// Pipeline:
//   K0  f32->bf16 convert of feat
//   K1  transpose+convert q_w / k_w / conv_w to K-major bf16 (B-fragment friendly)
//   K2  logw[b,q,g,k] = log(max(relu(posemb(b,q,k)·fc1_w[g]+fc1_b[g]),1e-6))  (VALU/trans)
//   K3  bf16 WMMA GEMM, 16x(NT*16) tile per wave (A-fragment reuse across NT B tiles):
//         q  = bf16(feat @ q_wT + q_b)        (3200x1024x1024, NT=4)
//         k  = bf16(feat @ k_wT + k_b)        (3200x1024x1024, NT=4)
//         kfeat[b,g] = bf16(feat[b] @ conv_w[g]T)   (100x32x1024, batched, NT=2)
//   K4  fused attention: aff = q·kT/sqrt(32) (1 WMMA/tile), +logw, mask, softmax,
//       sm @ kfeat (WMMA, K padded to 128), + conv_b + feat residual, ReLU.
// ---------------------------------------------------------------------------

#define DEVINL __device__ __forceinline__

typedef __attribute__((ext_vector_type(16))) __bf16 v16bf;
typedef __attribute__((ext_vector_type(8)))  float  v8f;

union FragBF { unsigned u[8]; v16bf v; };

DEVINL unsigned short f2bf(float f) {            // round-to-nearest-even f32->bf16
  unsigned u = __float_as_uint(f);
  u += 0x7FFFu + ((u >> 16) & 1u);
  return (unsigned short)(u >> 16);
}

static constexpr int Bb = 32, Nn = 100, Gg = 32, Dd = 1024, DG = 32;

// --------------------------- K0: elementwise convert ------------------------
__global__ void cvt_bf16_kernel(const float* __restrict__ in,
                                unsigned short* __restrict__ out, int n) {
  int i = blockIdx.x * blockDim.x + threadIdx.x;
  if (i < n) out[i] = f2bf(in[i]);
}

// ------------------- K1: batched transpose to K-major bf16 ------------------
// in: (batch, R, C) f32 ; out: (batch, C, R) bf16  -> out[b][c][r] = in[b][r][c]
__global__ void transpose_bf16_kernel(const float* __restrict__ in,
                                      unsigned short* __restrict__ out,
                                      int batch, int R, int C) {
  int i = blockIdx.x * blockDim.x + threadIdx.x;
  int total = batch * R * C;
  if (i >= total) return;
  int bz = i / (R * C);
  int rem = i - bz * (R * C);
  int c = rem / R;
  int r = rem - c * R;
  out[i] = f2bf(in[(size_t)bz * R * C + (size_t)r * C + c]);
}

// ------------------------------ K2: logw ------------------------------------
// logw[b,q,g,k] = log(max(relu(emb(b,q,k)·fc1_w[g] + fc1_b[g]), 1e-6))
__global__ void logw_kernel(const float* __restrict__ bbox,
                            const float* __restrict__ fc1_w,
                            const float* __restrict__ fc1_b,
                            float* __restrict__ logw) {
  __shared__ float fw[Gg * 64];
  __shared__ float fb[Gg];
  for (int i = threadIdx.x; i < Gg * 64; i += blockDim.x) fw[i] = fc1_w[i];
  if (threadIdx.x < Gg) fb[threadIdx.x] = fc1_b[threadIdx.x];
  __syncthreads();

  int t = blockIdx.x * blockDim.x + threadIdx.x;   // exactly B*N*N threads
  int b = t / (Nn * Nn);
  int r = t - b * (Nn * Nn);
  int q  = r / Nn;
  int kk = r - q * Nn;

  const float* bq = bbox + (size_t)(b * Nn + q) * 4;
  const float* bk = bbox + (size_t)(b * Nn + kk) * 4;
  float xq0 = bq[0], yq0 = bq[1], xq1 = bq[2], yq1 = bq[3];
  float xk0 = bk[0], yk0 = bk[1], xk1 = bk[2], yk1 = bk[3];
  float bwq = xq1 - xq0 + 1.0f, bhq = yq1 - yq0 + 1.0f;
  float bwk = xk1 - xk0 + 1.0f, bhk = yk1 - yk0 + 1.0f;
  float cxq = 0.5f * (xq0 + xq1), cyq = 0.5f * (yq0 + yq1);
  float cxk = 0.5f * (xk0 + xk1), cyk = 0.5f * (yk0 + yk1);

  float pos[4];
  pos[0] = __logf(fmaxf(fabsf((cxq - cxk) / bwq), 1e-3f));
  pos[1] = __logf(fmaxf(fabsf((cyq - cyk) / bhq), 1e-3f));
  pos[2] = __logf(bwq / bwk);
  pos[3] = __logf(bhq / bhk);

  const float dim[8] = {1.0f, 2.37137371f, 5.62341325f, 13.3352143f,
                        31.6227766f, 74.9894209f, 177.827941f, 421.696503f};
  float emb[64];
#pragma unroll
  for (int p = 0; p < 4; ++p) {
#pragma unroll
    for (int f = 0; f < 8; ++f) {
      float dv = 100.0f * pos[p] / dim[f];
      emb[p * 16 + f]     = __sinf(dv);
      emb[p * 16 + 8 + f] = __cosf(dv);
    }
  }
  float* outp = logw + (((size_t)(b * Nn + q) * Gg) * Nn) + kk;
  for (int g = 0; g < Gg; ++g) {
    float s = fb[g];
    const float* w = &fw[g * 64];
#pragma unroll
    for (int f = 0; f < 64; ++f) s += emb[f] * w[f];
    s = fmaxf(s, 0.0f);                       // relu
    outp[(size_t)g * Nn] = __logf(fmaxf(s, 1e-6f));
  }
}

// --------------------- K3: bf16 WMMA GEMM, 16 x (NT*16) per wave ------------
// C[m,n] = bf16( sum_k A[m,k]*B[k,n] + bias[n] )
// A: row-major bf16 (lda), B: K-major bf16 (ldb), C: bf16 (ldc).
// Batched: aIdx = z / zB, bIdx = z % zB, C batch = z.
template <int NT>
__global__ void gemm_bf16_wmma(const unsigned short* __restrict__ A,
                               const unsigned short* __restrict__ Bm,
                               const float* __restrict__ bias,
                               unsigned short* __restrict__ C,
                               int M, int N, int K,
                               int lda, int ldb, int ldc,
                               long a_zstride, long b_zstride, long c_zstride,
                               int zB) {
  int z = blockIdx.z;
  A  += (long)(z / zB) * a_zstride;
  Bm += (long)(z % zB) * b_zstride;
  C  += (long)z * c_zstride;

  int lane = threadIdx.x & 31;
  int wave = threadIdx.x >> 5;
  int groups_n = N / (16 * NT);                    // N-tile groups of NT
  int tiles_m = (M + 15) >> 4;
  int tid = blockIdx.x * (blockDim.x >> 5) + wave;
  if (tid >= tiles_m * groups_n) return;           // wave-uniform exit (EXEC stays all-1)
  int tm = tid / groups_n, gn = tid - tm * groups_n;
  int ncol0 = gn * (16 * NT);

  int half = lane >> 4, l16 = lane & 15;
  int row = tm * 16 + l16;
  int rowc = row < M ? row : M - 1;                // clamp A reads, guard stores

  v8f acc[NT];
#pragma unroll
  for (int nt = 0; nt < NT; ++nt) acc[nt] = (v8f){};

  for (int kk = 0; kk < K; kk += 32) {
    FragBF a;
#pragma unroll
    for (int j = 0; j < 8; ++j) {                  // A: 16-bit swizzle (ISA 7.12.2)
      int kb = kk + ((j < 4) ? (half * 8 + 2 * j) : (16 + half * 8 + 2 * (j - 4)));
      a.u[j] = *(const unsigned*)(A + (size_t)rowc * lda + kb);
    }
    int c0 = kk + half * 8;                        // B: lane = N col, same K swizzle as A
    const unsigned short* brow = Bm + (size_t)c0 * ldb + ncol0 + l16;
#pragma unroll
    for (int nt = 0; nt < NT; ++nt) {              // A-fragment reused NT times
      FragBF bfrag;
#pragma unroll
      for (int j = 0; j < 8; ++j) {                // VGPR j = K pair (kr, kr+1)
        int kr = (j < 4) ? (2 * j) : (16 + 2 * (j - 4));
        bfrag.u[j] = (unsigned)brow[(size_t)kr * ldb + nt * 16] |
                     ((unsigned)brow[(size_t)(kr + 1) * ldb + nt * 16] << 16);
      }
      acc[nt] = __builtin_amdgcn_wmma_f32_16x16x32_bf16(false, a.v, false, bfrag.v,
                                                        (short)0, acc[nt], false, false);
    }
  }
#pragma unroll
  for (int nt = 0; nt < NT; ++nt) {
    int ncol = ncol0 + nt * 16 + l16;
    float bv = bias ? bias[ncol] : 0.0f;
#pragma unroll
    for (int v = 0; v < 8; ++v) {                  // C: VGPR=row, lane=col
      int m = tm * 16 + v + half * 8;
      if (m < M) C[(size_t)m * ldc + ncol] = f2bf(acc[nt][v] + bv);
    }
  }
}

// --------------------- K4: fused attention core -----------------------------
// One wave per (q-tile, g, b). aff WMMA + softmax + sm@kfeat WMMA + epilogue.
__global__ void __launch_bounds__(32)
attn_kernel(const unsigned short* __restrict__ qbf,
            const unsigned short* __restrict__ kbf,
            const unsigned short* __restrict__ kfeat,
            const float* __restrict__ logw,
            const unsigned char* __restrict__ mask,
            const float* __restrict__ feat,
            const float* __restrict__ conv_b,
            float* __restrict__ out) {
  const int qt = blockIdx.x;   // 0..6
  const int g  = blockIdx.y;   // 0..31
  const int b  = blockIdx.z;   // 0..31
  const int lane = threadIdx.x;
  const int half = lane >> 4, l16 = lane & 15;

  __shared__ unsigned short kT[32][114];   // k transposed: [d][k_], padded
  __shared__ float smrow[16][128];         // softmax rows, K padded to 128

  // Stage k_g transposed into LDS: global reads are contiguous b32 per k row.
#pragma unroll 4
  for (int it = 0; it < 56; ++it) {
    int kr = it * 2 + half;                       // 0..111
    int kc = kr < Nn ? kr : Nn - 1;
    unsigned u = *(const unsigned*)(kbf + ((size_t)(b * Nn + kc) * Dd) + g * DG + 2 * l16);
    kT[2 * l16][kr]     = (unsigned short)(u & 0xffffu);
    kT[2 * l16 + 1][kr] = (unsigned short)(u >> 16);
  }
  __syncthreads();

  // A fragment: q tile (16 q-rows x DG=32), K = DG -> single WMMA per N tile.
  int qrow = qt * 16 + l16;
  int qrc = qrow < Nn ? qrow : Nn - 1;
  FragBF afrag;
#pragma unroll
  for (int j = 0; j < 8; ++j) {
    int kb = (j < 4) ? (half * 8 + 2 * j) : (16 + half * 8 + 2 * (j - 4));
    afrag.u[j] = *(const unsigned*)(qbf + (size_t)(b * Nn + qrc) * Dd + g * DG + kb);
  }

  v8f acc[7];
#pragma unroll
  for (int nt = 0; nt < 7; ++nt) {
    FragBF bfrag;
    int c = nt * 16 + l16;                        // key col
#pragma unroll
    for (int j = 0; j < 8; ++j) {
      int kb = (j < 4) ? (half * 8 + 2 * j) : (16 + half * 8 + 2 * (j - 4));
      bfrag.u[j] = (unsigned)kT[kb][c] | ((unsigned)kT[kb + 1][c] << 16);
    }
    v8f z = {};
    acc[nt] = __builtin_amdgcn_wmma_f32_16x16x32_bf16(false, afrag.v, false, bfrag.v,
                                                      (short)0, z, false, false);
  }

  // weighted = logw + aff/sqrt(DG); masked -> -1e9; invalid cols -> -3e38
  const float scale = 0.17677669529663687f;       // 1/sqrt(32)
#pragma unroll
  for (int nt = 0; nt < 7; ++nt) {
#pragma unroll
    for (int v = 0; v < 8; ++v) {
      int col = nt * 16 + l16;
      int row = qt * 16 + v + 8 * half;
      float w = -3.0e38f;
      if (col < Nn && row < Nn) {
        size_t idx = (((size_t)(b * Nn + row) * Gg) + g) * Nn + col;
        w = logw[idx] + acc[nt][v] * scale;
        if (mask[idx]) w = -1.0e9f;
      }
      acc[nt][v] = w;
    }
  }

  // Softmax over k (the N dim of the tile row-block): reduce across the 16
  // lanes of each half (C layout: N = lane&15) and across the 7 N-tiles.
#pragma unroll
  for (int v = 0; v < 8; ++v) {
    float m = -3.0e38f;
#pragma unroll
    for (int nt = 0; nt < 7; ++nt) m = fmaxf(m, acc[nt][v]);
#pragma unroll
    for (int xm = 1; xm < 16; xm <<= 1) m = fmaxf(m, __shfl_xor(m, xm, 32));
    float s = 0.0f;
    float e[7];
#pragma unroll
    for (int nt = 0; nt < 7; ++nt) { e[nt] = __expf(acc[nt][v] - m); s += e[nt]; }
#pragma unroll
    for (int xm = 1; xm < 16; xm <<= 1) s += __shfl_xor(s, xm, 32);
    float inv = 1.0f / s;
    int rloc = v + 8 * half;
#pragma unroll
    for (int nt = 0; nt < 7; ++nt) {
      int col = nt * 16 + l16;
      smrow[rloc][col] = (col < Nn) ? e[nt] * inv : 0.0f;
    }
  }
  // zero padded K columns 112..127
#pragma unroll
  for (int i = 0; i < 8; ++i) {
    int idx = lane + i * 32;                      // 0..255
    smrow[idx >> 4][112 + (idx & 15)] = 0.0f;
  }
  __syncthreads();

  // lin = sm (16 x 128) @ kfeat[b,g] (128 x 32), K padded (sm zero past 100)
  v8f acc2[2] = {{}, {}};
#pragma unroll
  for (int ks = 0; ks < 4; ++ks) {
    FragBF a2;
#pragma unroll
    for (int j = 0; j < 8; ++j) {
      int p0 = ks * 32 + ((j < 4) ? (half * 8 + 2 * j) : (16 + half * 8 + 2 * (j - 4)));
      unsigned lo = f2bf(smrow[l16][p0]);
      unsigned hi = f2bf(smrow[l16][p0 + 1]);
      a2.u[j] = lo | (hi << 16);
    }
#pragma unroll
    for (int nt2 = 0; nt2 < 2; ++nt2) {
      FragBF b2;
      const unsigned short* kb2 = kfeat + (size_t)(b * Gg + g) * Nn * DG + nt2 * 16 + l16;
#pragma unroll
      for (int j = 0; j < 8; ++j) {
        int c = ks * 32 + ((j < 4) ? (half * 8 + 2 * j) : (16 + half * 8 + 2 * (j - 4)));
        int c0 = c < Nn ? c : Nn - 1;             // A is zero there anyway
        int c1 = c + 1 < Nn ? c + 1 : Nn - 1;
        b2.u[j] = (unsigned)kb2[(size_t)c0 * DG] | ((unsigned)kb2[(size_t)c1 * DG] << 16);
      }
      acc2[nt2] = __builtin_amdgcn_wmma_f32_16x16x32_bf16(false, a2.v, false, b2.v,
                                                          (short)0, acc2[nt2], false, false);
    }
  }

  // epilogue: + conv_b + residual, ReLU
#pragma unroll
  for (int nt2 = 0; nt2 < 2; ++nt2) {
    int o = nt2 * 16 + l16;
    int d = g * DG + o;
    float cb = conv_b[d];
#pragma unroll
    for (int v = 0; v < 8; ++v) {
      int row = qt * 16 + v + 8 * half;
      if (row < Nn) {
        size_t idx = (size_t)(b * Nn + row) * Dd + d;
        out[idx] = fmaxf(feat[idx] + acc2[nt2][v] + cb, 0.0f);
      }
    }
  }
}

// ---------------------------------------------------------------------------
extern "C" void kernel_launch(void* const* d_in, const int* in_sizes, int n_in,
                              void* d_out, int out_size, void* d_ws, size_t ws_size,
                              hipStream_t stream) {
  (void)in_sizes; (void)n_in; (void)out_size; (void)ws_size;
  const float* feat    = (const float*)d_in[0];
  const float* bbox    = (const float*)d_in[1];
  const unsigned char* fmask = (const unsigned char*)d_in[2];  // bool mask
  const float* fc1_w   = (const float*)d_in[3];
  const float* fc1_b   = (const float*)d_in[4];
  const float* q_w     = (const float*)d_in[5];
  const float* q_b     = (const float*)d_in[6];
  const float* k_w     = (const float*)d_in[7];
  const float* k_b     = (const float*)d_in[8];
  const float* conv_w  = (const float*)d_in[9];
  const float* conv_b  = (const float*)d_in[10];
  float* outp = (float*)d_out;

  // workspace layout (bytes)
  char* w = (char*)d_ws;
  unsigned short* feat_bf = (unsigned short*)(w);                 // 6,553,600
  unsigned short* qwT_bf  = (unsigned short*)(w + 6553600);       // 2,097,152
  unsigned short* kwT_bf  = (unsigned short*)(w + 8650752);       // 2,097,152
  unsigned short* cvT_bf  = (unsigned short*)(w + 10747904);      // 2,097,152
  unsigned short* q_bf    = (unsigned short*)(w + 12845056);      // 6,553,600
  unsigned short* k_bf    = (unsigned short*)(w + 19398656);      // 6,553,600
  unsigned short* kf_bf   = (unsigned short*)(w + 25952256);      // 6,553,600
  float*          logw    = (float*)(w + 32505856);               // 40,960,000

  // K0: feat -> bf16
  {
    int n = Bb * Nn * Dd;
    cvt_bf16_kernel<<<(n + 255) / 256, 256, 0, stream>>>(feat, feat_bf, n);
  }
  // K1: weights -> K-major bf16
  transpose_bf16_kernel<<<(Dd * Dd + 255) / 256, 256, 0, stream>>>(q_w, qwT_bf, 1, Dd, Dd);
  transpose_bf16_kernel<<<(Dd * Dd + 255) / 256, 256, 0, stream>>>(k_w, kwT_bf, 1, Dd, Dd);
  transpose_bf16_kernel<<<(Gg * DG * Dd + 255) / 256, 256, 0, stream>>>(conv_w, cvT_bf, Gg, DG, Dd);

  // K2: logw (B*N*N = 320000 threads exactly)
  logw_kernel<<<1250, 256, 0, stream>>>(bbox, fc1_w, fc1_b, logw);

  // K3a/b: q and k projections (M=3200, N=1024, K=1024)
  //   16x64 tile per wave: 200 * 16 = 3200 waves -> 400 blocks of 8 waves
  gemm_bf16_wmma<4><<<dim3(400, 1, 1), 256, 0, stream>>>(
      feat_bf, qwT_bf, q_b, q_bf, Bb * Nn, Dd, Dd, Dd, Dd, Dd, 0, 0, 0, 1);
  gemm_bf16_wmma<4><<<dim3(400, 1, 1), 256, 0, stream>>>(
      feat_bf, kwT_bf, k_b, k_bf, Bb * Nn, Dd, Dd, Dd, Dd, Dd, 0, 0, 0, 1);
  // K3c: kfeat[b,g] = feat[b] @ conv_w[g]^T  (M=100, N=32, K=1024), z = b*32+g
  //   16x32 tile per wave: 7 waves per z -> 1 block of 8 waves
  gemm_bf16_wmma<2><<<dim3(1, 1, Bb * Gg), 256, 0, stream>>>(
      feat_bf, cvT_bf, nullptr, kf_bf, Nn, DG, Dd, Dd, DG, DG,
      (long)Nn * Dd, (long)Dd * DG, (long)Nn * DG, Gg);

  // K4: fused attention
  attn_kernel<<<dim3(7, Gg, Bb), 32, 0, stream>>>(
      q_bf, k_bf, kf_bf, logw, fmask, feat, conv_b, outp);
}
